// Attention_61881888801211
// MI455X (gfx1250) — matrix operand order repaired and checked by reading, not compiled, not run on hardware
//
#include <hip/hip_runtime.h>
#include <hip/hip_bf16.h>
#include <math.h>

typedef __attribute__((ext_vector_type(16))) __bf16 v16bf;
typedef __attribute__((ext_vector_type(8)))  float  v8f;
typedef __attribute__((ext_vector_type(4)))  float  v4f;
typedef __attribute__((ext_vector_type(4)))  unsigned int v4u;

// ---- constants for this problem ----
#define BB   2
#define NN_  1600
#define NM   576
#define NS   1024
#define CC   768
#define HH   12
#define HD   64
#define NMEM 1600
#define LL   3200          // NMEM + NN_
#define RR   8
#define SCALING 0.125f     // 1/R
#define QSCALE  0.125f     // hd^-0.5 = 1/8

union FragBF { v4u u[2]; v16bf v; };

// Build A fragment (16-bit A 16x32 layout) from fp32 row: two 8-elem runs.
__device__ __forceinline__ v16bf fragA_f32(const float* __restrict__ Xr, int ks, int kbase) {
  v4f x0 = *(const v4f*)(Xr + ks + kbase);
  v4f x1 = *(const v4f*)(Xr + ks + kbase + 4);
  v4f x2 = *(const v4f*)(Xr + ks + kbase + 16);
  v4f x3 = *(const v4f*)(Xr + ks + kbase + 20);
  v16bf a;
#pragma unroll
  for (int j = 0; j < 4; ++j) {
    a[j]      = (__bf16)x0[j];
    a[4 + j]  = (__bf16)x1[j];
    a[8 + j]  = (__bf16)x2[j];
    a[12 + j] = (__bf16)x3[j];
  }
  return a;
}

// Build B fragment (16-bit B 32x16 layout) from fp32 row: two 8-elem runs.
__device__ __forceinline__ v16bf fragB_f32(const float* __restrict__ Wr, int ks, int kbase2) {
  v4f w0 = *(const v4f*)(Wr + ks + kbase2);
  v4f w1 = *(const v4f*)(Wr + ks + kbase2 + 4);
  v4f w2 = *(const v4f*)(Wr + ks + kbase2 + 16);
  v4f w3 = *(const v4f*)(Wr + ks + kbase2 + 20);
  v16bf b;
#pragma unroll
  for (int j = 0; j < 4; ++j) {
    b[j]      = (__bf16)w0[j];
    b[4 + j]  = (__bf16)w1[j];
    b[8 + j]  = (__bf16)w2[j];
    b[12 + j] = (__bf16)w3[j];
  }
  return b;
}

// Direct bf16 fragment loads (two 16B vector loads each).
__device__ __forceinline__ v16bf fragA_bf16(const __bf16* __restrict__ p, int ks, int kbase) {
  FragBF f;
  f.u[0] = *(const v4u*)(p + ks + kbase);
  f.u[1] = *(const v4u*)(p + ks + kbase + 16);
  return f.v;
}
__device__ __forceinline__ v16bf fragB_bf16(const __bf16* __restrict__ p, int ks, int kbase2) {
  FragBF f;
  f.u[0] = *(const v4u*)(p + ks + kbase2);
  f.u[1] = *(const v4u*)(p + ks + kbase2 + 16);
  return f.v;
}

// ---------- Generic WMMA GEMM: Out[M,Nc] = X[M,K] * W[Nc,K]^T + bias ----------
// Each wave computes a 16x64 strip: 4 accumulators sharing one A fragment.
__global__ __launch_bounds__(256)
void gemm_f32_bf16wmma_nt(const float* __restrict__ X, const float* __restrict__ W,
                          const float* __restrict__ bias, float* __restrict__ Out,
                          int M, int Nc, int K) {
  const int lane = threadIdx.x & 31;
  const int wv   = blockIdx.x * (blockDim.x >> 5) + (threadIdx.x >> 5);
  const int tilesN4 = Nc >> 6;                      // 64-col groups
  if (wv >= (M >> 4) * tilesN4) return;             // wave-uniform guard
  const int tm = wv / tilesN4, tn4 = wv % tilesN4;
  const int r15 = lane & 15;
  const int kbase  = (lane >> 4) << 3;              // A: 0 or 8
  const int kbase2 = (lane >> 4) << 3;              // B: 0 or 8
  const float* Xr = X + (size_t)(tm * 16 + r15) * K;
  const float* Wr0 = W + (size_t)(tn4 * 64 + r15) * K;
  const float* Wr1 = Wr0 + (size_t)16 * K;
  const float* Wr2 = Wr0 + (size_t)32 * K;
  const float* Wr3 = Wr0 + (size_t)48 * K;
  v8f acc0 = {0.f,0.f,0.f,0.f,0.f,0.f,0.f,0.f};
  v8f acc1 = acc0, acc2 = acc0, acc3 = acc0;
  for (int ks = 0; ks < K; ks += 32) {
    __builtin_prefetch(Xr + ks + 64, 0, 1);
    v16bf a  = fragA_f32(Xr, ks, kbase);
    v16bf b0 = fragB_f32(Wr0, ks, kbase2);
    v16bf b1 = fragB_f32(Wr1, ks, kbase2);
    v16bf b2 = fragB_f32(Wr2, ks, kbase2);
    v16bf b3 = fragB_f32(Wr3, ks, kbase2);
    acc0 = __builtin_amdgcn_wmma_f32_16x16x32_bf16(false, a, false, b0, (short)0, acc0, false, false);
    acc1 = __builtin_amdgcn_wmma_f32_16x16x32_bf16(false, a, false, b1, (short)0, acc1, false, false);
    acc2 = __builtin_amdgcn_wmma_f32_16x16x32_bf16(false, a, false, b2, (short)0, acc2, false, false);
    acc3 = __builtin_amdgcn_wmma_f32_16x16x32_bf16(false, a, false, b3, (short)0, acc3, false, false);
  }
  const int rhalf = (lane >> 4) << 3;
  v8f accs[4] = {acc0, acc1, acc2, acc3};
#pragma unroll
  for (int s = 0; s < 4; ++s) {
    const int col = tn4 * 64 + s * 16 + r15;
    const float bv = bias ? bias[col] : 0.0f;
#pragma unroll
    for (int r = 0; r < 8; ++r) {
      int row = tm * 16 + r + rhalf;
      Out[(size_t)row * Nc + col] = accs[s][r] + bv;
    }
  }
}

// ---------- QKV LoRA (3 groups, rank 8), adds into qkv buffer ----------
__global__ __launch_bounds__(256)
void lora_qkv_add(const float* __restrict__ x, const float* __restrict__ A,
                  const float* __restrict__ Bm, float* __restrict__ qkv) {
  __shared__ float xrow[CC];
  __shared__ float z[24];
  const int tid = threadIdx.x;
  const size_t row = blockIdx.x;
  const float* xr = x + row * CC;
  for (int i = tid; i < CC; i += 256) xrow[i] = xr[i];
  __syncthreads();
  if (tid < 24) {
    const float* Ar = A + (size_t)tid * CC;          // (g*R+r)*C
    float s = 0.f;
    for (int c = 0; c < CC; ++c) s += xrow[c] * Ar[c];
    z[tid] = s;
  }
  __syncthreads();
  float* qr = qkv + row * (size_t)(3 * CC);
  for (int o = tid; o < 3 * CC; o += 256) {
    int g = o / CC, c = o % CC;
    const float* Br = Bm + ((size_t)g * CC + c) * RR;
    float acc = 0.f;
#pragma unroll
    for (int r = 0; r < RR; ++r) acc += z[g * RR + r] * Br[r];
    qr[o] += SCALING * acc;
  }
}

// ---------- ID_K: (B,Nm,H) = id_total@Wk^T + b + 1/R * (id@A^T)@B^T  (E=1 => rw=1) ----
__global__ __launch_bounds__(64)
void idk_kernel(const float* __restrict__ xid, const float* __restrict__ Wk,
                const float* __restrict__ bk, const float* __restrict__ Ak,
                const float* __restrict__ Bk, float* __restrict__ idk) {
  __shared__ float xrow[CC];
  __shared__ float z[RR];
  __shared__ float base[HH];
  const int tid = threadIdx.x;
  const size_t row = blockIdx.x;                     // b*NM + n
  const float* xr = xid + row * CC;
  for (int i = tid; i < CC; i += 64) xrow[i] = xr[i];
  __syncthreads();
  if (tid < RR) {
    const float* Ar = Ak + (size_t)tid * CC;
    float s = 0.f;
    for (int c = 0; c < CC; ++c) s += xrow[c] * Ar[c];
    z[tid] = s;
  } else if (tid >= 32 && tid < 32 + HH) {
    int h = tid - 32;
    const float* Wr = Wk + (size_t)h * CC;
    float s = 0.f;
    for (int c = 0; c < CC; ++c) s += xrow[c] * Wr[c];
    base[h] = s + bk[h];
  }
  __syncthreads();
  if (tid < HH) {
    float acc = 0.f;
#pragma unroll
    for (int r = 0; r < RR; ++r) acc += z[r] * Bk[tid * RR + r];
    idk[row * HH + tid] = base[tid] + SCALING * acc;
  }
}

// ---------- ID_V LoRA add (E=1) ----------
__global__ __launch_bounds__(256)
void idv_lora_add(const float* __restrict__ xid, const float* __restrict__ Av,
                  const float* __restrict__ Bv, float* __restrict__ idv) {
  __shared__ float xrow[CC];
  __shared__ float z[RR];
  const int tid = threadIdx.x;
  const size_t row = blockIdx.x;
  const float* xr = xid + row * CC;
  for (int i = tid; i < CC; i += 256) xrow[i] = xr[i];
  __syncthreads();
  if (tid < RR) {
    const float* Ar = Av + (size_t)tid * CC;
    float s = 0.f;
    for (int c = 0; c < CC; ++c) s += xrow[c] * Ar[c];
    z[tid] = s;
  }
  __syncthreads();
  float* vr = idv + row * CC;
  for (int c = tid; c < CC; c += 256) {
    const float* Br = Bv + (size_t)c * RR;
    float acc = 0.f;
#pragma unroll
    for (int r = 0; r < RR; ++r) acc += z[r] * Br[r];
    vr[c] += SCALING * acc;
  }
}

// ---------- build Q/K/V (bf16 attention layout) + fp32 k_m/v_m outputs ----------
__global__ __launch_bounds__(256)
void build_kernel(const float* __restrict__ qkv, const float* __restrict__ idk,
                  const float* __restrict__ idv, const float* __restrict__ mem_k,
                  const float* __restrict__ mem_v,
                  __bf16* __restrict__ qbf, __bf16* __restrict__ kf, __bf16* __restrict__ vf,
                  float* __restrict__ km_out, float* __restrict__ vm_out) {
  long long idx = (long long)blockIdx.x * blockDim.x + threadIdx.x;
  const long long total = (long long)BB * HH * LL * HD;
  if (idx >= total) return;
  int d = (int)(idx & 63);
  long long t = idx >> 6;
  int j = (int)(t % LL); t /= LL;
  int h = (int)(t % HH);
  int b = (int)(t / HH);
  size_t o = (((size_t)(b * HH + h) * LL) + j) * HD + d;
  if (j < NMEM) {
    size_t mi = (((size_t)(b * HH + h) * NMEM) + j) * HD + d;
    kf[o] = (__bf16)mem_k[mi];
    vf[o] = (__bf16)mem_v[mi];
  } else {
    int n = j - NMEM;
    size_t base = ((size_t)(b * NN_ + n) * 3) * CC + h * HD + d;
    float qq = qkv[base] * QSCALE;
    float kk = qkv[base + CC];
    float vv = qkv[base + 2 * CC];
    qbf[(((size_t)(b * HH + h) * NN_) + n) * HD + d] = (__bf16)qq;
    if (n < NM) {
      float gate = 1.f + tanhf(idk[((size_t)b * NM + n) * HH + h]);
      float km = kk * gate;
      float vm = vv + idv[((size_t)b * NM + n) * CC + h * HD + d];
      size_t mo = (((size_t)(b * HH + h) * NM) + n) * HD + d;
      km_out[mo] = km;
      vm_out[mo] = vm;
      kf[o] = (__bf16)km;
      vf[o] = (__bf16)vm;
    } else {
      kf[o] = (__bf16)kk;
      vf[o] = (__bf16)vv;
    }
  }
}

// ---------- batched attention scores: S = Q * K^T (bf16 WMMA, 16x64 per wave) ----------
__global__ __launch_bounds__(256)
void attn_score(const __bf16* __restrict__ Q, const __bf16* __restrict__ Kf,
                float* __restrict__ S, int M, int Nk, int qRowOff, int kRowOff,
                size_t strideQ, size_t strideK, size_t strideS) {
  const int bh = blockIdx.y;
  const int lane = threadIdx.x & 31;
  const int wv = blockIdx.x * (blockDim.x >> 5) + (threadIdx.x >> 5);
  const int tilesN4 = Nk >> 6;
  if (wv >= (M >> 4) * tilesN4) return;
  const int tm = wv / tilesN4, tn4 = wv % tilesN4;
  const int r15 = lane & 15;
  const int kbase  = (lane >> 4) << 3;
  const int kbase2 = (lane >> 4) << 3;
  const __bf16* Qr = Q + (size_t)bh * strideQ + (size_t)(qRowOff + tm * 16 + r15) * HD;
  const __bf16* Kr0 = Kf + (size_t)bh * strideK + (size_t)(kRowOff + tn4 * 64 + r15) * HD;
  const __bf16* Kr1 = Kr0 + (size_t)16 * HD;
  const __bf16* Kr2 = Kr0 + (size_t)32 * HD;
  const __bf16* Kr3 = Kr0 + (size_t)48 * HD;
  v8f acc0 = {0.f,0.f,0.f,0.f,0.f,0.f,0.f,0.f};
  v8f acc1 = acc0, acc2 = acc0, acc3 = acc0;
#pragma unroll
  for (int ks = 0; ks < HD; ks += 32) {
    v16bf a  = fragA_bf16(Qr, ks, kbase);
    v16bf b0 = fragB_bf16(Kr0, ks, kbase2);
    v16bf b1 = fragB_bf16(Kr1, ks, kbase2);
    v16bf b2 = fragB_bf16(Kr2, ks, kbase2);
    v16bf b3 = fragB_bf16(Kr3, ks, kbase2);
    acc0 = __builtin_amdgcn_wmma_f32_16x16x32_bf16(false, a, false, b0, (short)0, acc0, false, false);
    acc1 = __builtin_amdgcn_wmma_f32_16x16x32_bf16(false, a, false, b1, (short)0, acc1, false, false);
    acc2 = __builtin_amdgcn_wmma_f32_16x16x32_bf16(false, a, false, b2, (short)0, acc2, false, false);
    acc3 = __builtin_amdgcn_wmma_f32_16x16x32_bf16(false, a, false, b3, (short)0, acc3, false, false);
  }
  float* Sr = S + (size_t)bh * strideS;
  const int rhalf = (lane >> 4) << 3;
  v8f accs[4] = {acc0, acc1, acc2, acc3};
#pragma unroll
  for (int s = 0; s < 4; ++s) {
    const int col = tn4 * 64 + s * 16 + r15;
#pragma unroll
    for (int r = 0; r < 8; ++r)
      Sr[(size_t)(tm * 16 + r + rhalf) * Nk + col] = accs[s][r];
  }
}

// ---------- exact top-k softmax per row (radix binary search on float keys) ----------
__global__ __launch_bounds__(256)
void topk_softmax_rows(const float* __restrict__ S, __bf16* __restrict__ P, int Lr, int k) {
  extern __shared__ float vals[];
  __shared__ float    redf[256];
  __shared__ unsigned redu[256];
  const int tid = threadIdx.x;
  const size_t row = blockIdx.x;
  const float* Sr = S + row * (size_t)Lr;
  float lmax = -3.4e38f;
  for (int i = tid; i < Lr; i += 256) { float v = Sr[i]; vals[i] = v; lmax = fmaxf(lmax, v); }
  redf[tid] = lmax; __syncthreads();
  for (int s = 128; s > 0; s >>= 1) { if (tid < s) redf[tid] = fmaxf(redf[tid], redf[tid + s]); __syncthreads(); }
  const float rowmax = redf[0];
  __syncthreads();
  unsigned prefix = 0u;
  for (int bit = 31; bit >= 0; --bit) {
    unsigned cand = prefix | (1u << bit);
    unsigned cnt = 0;
    for (int i = tid; i < Lr; i += 256) {
      unsigned u = __float_as_uint(vals[i]);
      unsigned key = (u & 0x80000000u) ? ~u : (u | 0x80000000u);
      cnt += (key >= cand) ? 1u : 0u;
    }
    redu[tid] = cnt; __syncthreads();
    for (int s = 128; s > 0; s >>= 1) { if (tid < s) redu[tid] += redu[tid + s]; __syncthreads(); }
    if (redu[0] >= (unsigned)k) prefix = cand;
    __syncthreads();
  }
  float lsum = 0.f;
  for (int i = tid; i < Lr; i += 256) {
    float v = vals[i];
    unsigned u = __float_as_uint(v);
    unsigned key = (u & 0x80000000u) ? ~u : (u | 0x80000000u);
    if (key >= prefix) lsum += __expf(v - rowmax);
  }
  redf[tid] = lsum; __syncthreads();
  for (int s = 128; s > 0; s >>= 1) { if (tid < s) redf[tid] += redf[tid + s]; __syncthreads(); }
  const float inv = 1.0f / redf[0];
  __bf16* Pr = P + row * (size_t)Lr;
  for (int i = tid; i < Lr; i += 256) {
    float v = vals[i];
    unsigned u = __float_as_uint(v);
    unsigned key = (u & 0x80000000u) ? ~u : (u | 0x80000000u);
    Pr[i] = (__bf16)((key >= prefix) ? __expf(v - rowmax) * inv : 0.0f);
  }
}

// ---------- batched PV: Out(b,n,h*64+d) = P * V (one wave covers all 4 d-subtiles) ----
__global__ __launch_bounds__(256)
void attn_pv(const __bf16* __restrict__ P, const __bf16* __restrict__ V,
             float* __restrict__ Out, int M, int Kd, int vRowOff,
             size_t strideP, size_t strideV, int rowOutOff) {
  const int bh = blockIdx.y;
  const int b = bh / HH, h = bh % HH;
  const int lane = threadIdx.x & 31;
  const int wv = blockIdx.x * (blockDim.x >> 5) + (threadIdx.x >> 5);
  if (wv >= (M >> 4)) return;
  const int tm = wv;
  const int r15 = lane & 15;
  const int kbase  = (lane >> 4) << 3;
  const int kbase2 = (lane >> 4) << 4;
  const __bf16* Pr = P + (size_t)bh * strideP + (size_t)(tm * 16 + r15) * Kd;
  const __bf16* Vb = V + (size_t)bh * strideV + (size_t)vRowOff * HD;
  v8f acc0 = {0.f,0.f,0.f,0.f,0.f,0.f,0.f,0.f};
  v8f acc1 = acc0, acc2 = acc0, acc3 = acc0;
  for (int ks = 0; ks < Kd; ks += 32) {
    v16bf a = fragA_bf16(Pr, ks, kbase);
    v16bf b0, b1, b2, b3;
#pragma unroll
    for (int j = 0; j < 16; ++j) {
      const __bf16* vrow = Vb + (size_t)(ks + kbase + (j & 7) + ((j >> 3) << 4)) * HD;
      b0[j] = vrow[r15];
      b1[j] = vrow[16 + r15];
      b2[j] = vrow[32 + r15];
      b3[j] = vrow[48 + r15];
    }
    acc0 = __builtin_amdgcn_wmma_f32_16x16x32_bf16(false, a, false, b0, (short)0, acc0, false, false);
    acc1 = __builtin_amdgcn_wmma_f32_16x16x32_bf16(false, a, false, b1, (short)0, acc1, false, false);
    acc2 = __builtin_amdgcn_wmma_f32_16x16x32_bf16(false, a, false, b2, (short)0, acc2, false, false);
    acc3 = __builtin_amdgcn_wmma_f32_16x16x32_bf16(false, a, false, b3, (short)0, acc3, false, false);
  }
  const int rhalf = (lane >> 4) << 3;
  v8f accs[4] = {acc0, acc1, acc2, acc3};
#pragma unroll
  for (int s = 0; s < 4; ++s) {
    const int col = s * 16 + r15;
#pragma unroll
    for (int r = 0; r < 8; ++r) {
      int m = tm * 16 + r + rhalf;
      Out[((size_t)(b * NN_ + rowOutOff + m)) * CC + h * HD + col] = accs[s][r];
    }
  }
}

// ---------- proj routed LoRA (E=4, softmax routing), adds into out ----------
__global__ __launch_bounds__(256)
void proj_lora_add(const float* __restrict__ X, const float* __restrict__ routeW,
                   const float* __restrict__ A, const float* __restrict__ Bm,
                   float* __restrict__ out) {
  __shared__ float xrow[CC];
  __shared__ float z[32];
  __shared__ float logits[4];
  __shared__ float rw[4];
  const int tid = threadIdx.x;
  const size_t row = blockIdx.x;
  const float* xr = X + row * CC;
  for (int i = tid; i < CC; i += 256) xrow[i] = xr[i];
  __syncthreads();
  if (tid < 32) {
    const float* Ar = A + (size_t)tid * CC;          // (e*R+r)*C
    float s = 0.f;
    for (int c = 0; c < CC; ++c) s += xrow[c] * Ar[c];
    z[tid] = s;
  } else if (tid >= 32 && tid < 36) {
    int e = tid - 32;
    const float* Wr = routeW + (size_t)e * CC;
    float s = 0.f;
    for (int c = 0; c < CC; ++c) s += xrow[c] * Wr[c];
    logits[e] = s;
  }
  __syncthreads();
  if (tid == 0) {
    float m = fmaxf(fmaxf(logits[0], logits[1]), fmaxf(logits[2], logits[3]));
    float s = 0.f, e4[4];
#pragma unroll
    for (int e = 0; e < 4; ++e) { e4[e] = __expf(logits[e] - m); s += e4[e]; }
#pragma unroll
    for (int e = 0; e < 4; ++e) rw[e] = e4[e] / s;
  }
  __syncthreads();
  float* orow = out + row * CC;
  for (int c = tid; c < CC; c += 256) {
    float acc = 0.f;
#pragma unroll
    for (int e = 0; e < 4; ++e) {
      const float* Br = Bm + ((size_t)e * CC + c) * RR;
      float de = 0.f;
#pragma unroll
      for (int r = 0; r < RR; ++r) de += z[e * RR + r] * Br[r];
      acc += rw[e] * de;
    }
    orow[c] += SCALING * acc;
  }
}

extern "C" void kernel_launch(void* const* d_in, const int* in_sizes, int n_in,
                              void* d_out, int out_size, void* d_ws, size_t ws_size,
                              hipStream_t stream) {
  const float* x        = (const float*)d_in[0];
  const float* id_total = (const float*)d_in[1];
  const float* mem_k    = (const float*)d_in[2];
  const float* mem_v    = (const float*)d_in[3];
  const float* W_qkv    = (const float*)d_in[4];
  const float* b_qkv    = (const float*)d_in[5];
  const float* A_qkv    = (const float*)d_in[6];
  const float* B_qkv    = (const float*)d_in[7];
  const float* W_proj   = (const float*)d_in[8];
  const float* b_proj   = (const float*)d_in[9];
  const float* routeW_p = (const float*)d_in[10];
  const float* A_proj   = (const float*)d_in[11];
  const float* B_proj   = (const float*)d_in[12];
  const float* W_idk    = (const float*)d_in[13];
  const float* b_idk    = (const float*)d_in[14];
  const float* A_idk    = (const float*)d_in[16];
  const float* B_idk    = (const float*)d_in[17];
  const float* W_idv    = (const float*)d_in[18];
  const float* b_idv    = (const float*)d_in[19];
  const float* A_idv    = (const float*)d_in[21];
  const float* B_idv    = (const float*)d_in[22];

  float* out    = (float*)d_out;                   // (B,N,C)   = 2,457,600
  float* km_out = out + (size_t)BB * NN_ * CC;     // (B,H,Nm,hd) = 884,736
  float* vm_out = km_out + (size_t)BB * HH * NM * HD;

  // workspace layout (fp32 region then bf16 region)
  float* ws      = (float*)d_ws;
  float* qkv     = ws;                                   // 7,372,800
  float* idk     = qkv + (size_t)BB * NN_ * 3 * CC;      // 13,824
  float* idv     = idk + (size_t)BB * NM * HH;           // 884,736
  float* attnout = idv + (size_t)BB * NM * CC;           // 2,457,600
  float* Sbuf    = attnout + (size_t)BB * NN_ * CC;      // 78,643,200 (reused m/s)
  __bf16* qbf = (__bf16*)(Sbuf + (size_t)BB * HH * NS * LL);
  __bf16* kf  = qbf + (size_t)BB * HH * NN_ * HD;
  __bf16* vf  = kf + (size_t)BB * HH * LL * HD;
  __bf16* Pm  = vf + (size_t)BB * HH * LL * HD;
  __bf16* Ps  = Pm + (size_t)BB * HH * NM * NM;

  // 1) qkv = x @ W_qkv^T + b   [3200 x 2304 x 768] -> 200*36 = 7200 waves
  gemm_f32_bf16wmma_nt<<<900, 256, 0, stream>>>(x, W_qkv, b_qkv, qkv, BB * NN_, 3 * CC, CC);
  // 2) + merged LoRA
  lora_qkv_add<<<BB * NN_, 256, 0, stream>>>(x, A_qkv, B_qkv, qkv);
  // 3) ID_K
  idk_kernel<<<BB * NM, 64, 0, stream>>>(id_total, W_idk, b_idk, A_idk, B_idk, idk);
  // 4) ID_V = id_total @ W_idv^T + b   [1152 x 768 x 768] -> 72*12 = 864 waves
  gemm_f32_bf16wmma_nt<<<108, 256, 0, stream>>>(id_total, W_idv, b_idv, idv, BB * NM, CC, CC);
  idv_lora_add<<<BB * NM, 256, 0, stream>>>(id_total, A_idv, B_idv, idv);
  // 5) build gated K_m, V_m(+ID_V), bf16 Q/K_full/V_full; write km/vm outputs
  {
    long long total = (long long)BB * HH * LL * HD;  // 4,915,200
    build_kernel<<<(int)(total / 256), 256, 0, stream>>>(qkv, idk, idv, mem_k, mem_v,
                                                         qbf, kf, vf, km_out, vm_out);
  }
  const size_t sQ = (size_t)NN_ * HD, sK = (size_t)LL * HD;
  // 6) attn_m scores: Q rows 0..575 vs K_full rows NMEM..NMEM+575  (36*9 = 324 waves)
  {
    dim3 g(41, BB * HH);
    attn_score<<<g, 256, 0, stream>>>(qbf, kf, Sbuf, NM, NM, 0, NMEM, sQ, sK, (size_t)NM * NM);
  }
  topk_softmax_rows<<<BB * HH * NM, 256, NM * sizeof(float), stream>>>(Sbuf, Pm, NM, NM / 2);
  {
    dim3 g(5, BB * HH);   // 36 waves
    attn_pv<<<g, 256, 0, stream>>>(Pm, vf, attnout, NM, NM, NMEM, (size_t)NM * NM, sK, 0);
  }
  // 7) attn_s: Q rows 576..1599 vs full K (3200)  (64*50 = 3200 waves)
  {
    dim3 g(400, BB * HH);
    attn_score<<<g, 256, 0, stream>>>(qbf, kf, Sbuf, NS, LL, NM, 0, sQ, sK, (size_t)NS * LL);
  }
  topk_softmax_rows<<<BB * HH * NS, 256, LL * sizeof(float), stream>>>(Sbuf, Ps, LL, LL / 2);
  {
    dim3 g(8, BB * HH);   // 64 waves
    attn_pv<<<g, 256, 0, stream>>>(Ps, vf, attnout, NS, LL, 0, (size_t)NS * LL, sK, NM);
  }
  // 8) out = attnout @ W_proj^T + b_proj  [3200 x 768 x 768] -> 200*12 = 2400 waves
  gemm_f32_bf16wmma_nt<<<300, 256, 0, stream>>>(attnout, W_proj, b_proj, out, BB * NN_, CC, CC);
  // 9) + routed expert LoRA
  proj_lora_add<<<BB * NN_, 256, 0, stream>>>(attnout, routeW_p, A_proj, B_proj, out);
}
